// Decoder_11622181503142
// MI455X (gfx1250) — matrix operand order repaired and checked by reading, not compiled, not run on hardware
//
#include <hip/hip_runtime.h>

// ---------------------------------------------------------------------------
// GNN decoder for MI455X (gfx1250): bf16 WMMA (v_wmma_f32_16x16x32_bf16),
// fp32 accumulate + fp32 epilogues (LayerNorm / exact-erf GELU / bias).
//
// d_in layout assumption (setup_inputs dict order, params flattened in dict
// insertion order):
//   0 node_features (N*D f32)   1 edge_features (N*K*D f32)
//   2 neighbor_indices (N*K i32) 3 mask (N f32)
//   4 msg_w0 (L*D*4D) 5 msg_b0 (L*D) 6 msg_w1 (L*D*D) 7 msg_b1 (L*D)
//   8 msg_w2 (L*D*D)  9 msg_b2 (L*D) 10 ln1_w 11 ln1_b
//   12 dense_w0 (L*DF*D) 13 dense_b0 (L*DF) 14 dense_w1 (L*D*DF) 15 dense_b1
//   16 ln2_w 17 ln2_b
// ---------------------------------------------------------------------------

typedef __bf16 v16bf __attribute__((ext_vector_type(16)));
typedef float  v8f   __attribute__((ext_vector_type(8)));

constexpr int NN  = 40000;
constexpr int KK  = 48;
constexpr int DD  = 128;
constexpr int DFF = 512;
constexpr int LL  = 3;
constexpr int LDE = 136;   // padded bf16 stride for 128-wide tiles (272B: breaks 64-bank aliasing)
constexpr int LDH = 520;   // padded bf16 stride for 512-wide tile
constexpr float SCALE_INV = 1.0f / 30.0f;
constexpr float LN_EPS = 1e-5f;

__device__ __forceinline__ unsigned short f2bf(float f) {
  unsigned int u = __float_as_uint(f);
  u += 0x7FFFu + ((u >> 16) & 1u);          // round-to-nearest-even
  return (unsigned short)(u >> 16);
}
__device__ __forceinline__ float bf2f(unsigned short u) {
  return __uint_as_float(((unsigned int)u) << 16);
}
__device__ __forceinline__ float gelu_exact(float x) {
  return 0.5f * x * (1.0f + erff(x * 0.70710678118654752f));
}

union FragU { v16bf v; uint4 q[2]; };

// A fragment (16x32 bf16) from a row-major LDS tile.
// Layout: lanes 0-15 row M=l, K 0..15; lanes 16-31 K 16..31 (contiguous, matches B).
__device__ __forceinline__ v16bf load_a_lds(const unsigned short* base, int ld) {
  int l = threadIdx.x & 31;
  const unsigned short* p = base + (l & 15) * ld + ((l >> 4) << 4);
  FragU f;
  f.q[0] = *(const uint4*)(p);        // K = khalf .. khalf+7
  f.q[1] = *(const uint4*)(p + 8);    // K = khalf+8 .. khalf+15
  return f.v;
}

// B fragment (32x16 bf16) of W^T from row-major W[n][k] in global (bf16).
// Lane l -> column n = l&15; lanes<16 K 0..15, lanes>=16 K 16..31 (contiguous 32B/lane).
__device__ __forceinline__ v16bf load_b_gl(const unsigned short* W, int ld,
                                           int nbase, int kbase) {
  int l = threadIdx.x & 31;
  const unsigned short* p =
      W + (size_t)(nbase + (l & 15)) * ld + kbase + ((l >> 4) << 4);
  FragU f;
  f.q[0] = *(const uint4*)(p);
  f.q[1] = *(const uint4*)(p + 8);
  return f.v;
}

__device__ __forceinline__ v8f wmma_bf16(v16bf a, v16bf b, v8f c) {
  return __builtin_amdgcn_wmma_f32_16x16x32_bf16(false, a, false, b,
                                                 (short)0, c, false, false);
}

// ---------------------------------------------------------------------------
// Prep kernels
// ---------------------------------------------------------------------------
__global__ void k_cvt_bf16(const float* __restrict__ s,
                           unsigned short* __restrict__ d, int n) {
  int i = blockIdx.x * 256 + threadIdx.x;
  if (i < n) d[i] = f2bf(s[i]);
}
__global__ void k_copy_f32(const float* __restrict__ s,
                           float* __restrict__ d, int n) {
  int i = blockIdx.x * 256 + threadIdx.x;
  if (i < n) d[i] = s[i];
}

// ---------------------------------------------------------------------------
// Message kernel: one workgroup per node, 256 threads = 8 waves.
// agg[n] = sum_k msgMLP(h[n], edge[n,k], nodes[idx[n,k]]) / 30
// ---------------------------------------------------------------------------
__global__ __launch_bounds__(256)
void msg_kernel(const float* __restrict__ hbuf,
                const float* __restrict__ edge,
                const int* __restrict__ nbr,
                const unsigned short* __restrict__ nodes_bf,
                const unsigned short* __restrict__ W0,   // (128, 512) bf16
                const float* __restrict__ b0,
                const unsigned short* __restrict__ W1,   // (128, 128) bf16
                const float* __restrict__ b1,
                const unsigned short* __restrict__ W2,   // (128, 128) bf16
                const float* __restrict__ b2,
                float* __restrict__ agg) {
  __shared__ unsigned short Ebuf[KK * LDE];
  __shared__ unsigned short Gbuf[KK * LDE];
  __shared__ unsigned short M0buf[KK * LDE];
  __shared__ unsigned short M1buf[KK * LDE];
  __shared__ float hw0[DD];
  __shared__ float sums[256];

  const int n    = blockIdx.x;
  const int t    = threadIdx.x;
  const int lane = t & 31;
  const int wv   = t >> 5;

  // Edge tile: f32 -> bf16 into LDS (coalesced float4)
  const float4* eg = (const float4*)(edge + (size_t)n * KK * DD);
  for (int i = t; i < KK * DD / 4; i += 256) {
    float4 v = eg[i];
    int row = i >> 5, col = (i & 31) << 2;
    unsigned short* p = &Ebuf[row * LDE + col];
    p[0] = f2bf(v.x); p[1] = f2bf(v.y); p[2] = f2bf(v.z); p[3] = f2bf(v.w);
  }
  // Gather neighbor rows from the bf16 node table (L2-resident)
  for (int c = t; c < KK * 16; c += 256) {
    int row = c >> 4, part = c & 15;
    int id = nbr[n * KK + row];
    const uint4* src = (const uint4*)(nodes_bf + (size_t)id * DD) + part;
    *(uint4*)&Gbuf[row * LDE + part * 8] = *src;
  }
  // hw0 = b0 + h @ W0[:, 0:128]^T  (broadcast across the K edges)
  if (t < DD) {
    const float* hp = hbuf + (size_t)n * DD;
    const unsigned short* wr = W0 + (size_t)t * (4 * DD);
    float a = b0[t];
    for (int c = 0; c < DD; ++c) a += hp[c] * bf2f(wr[c]);
    hw0[t] = a;
  }
  __syncthreads();

  const int nt = wv;                   // each wave owns one 16-col tile
  const int colB = nt * 16 + (lane & 15);
  const int mo   = (lane >> 4) << 3;

  // m0 = gelu(E @ W0[:,128:256]^T + G @ W0[:,384:512]^T + hw0)
  {
    v8f a0 = {}, a1 = {}, a2 = {};
    for (int kk = 0; kk < 4; ++kk) {
      v16bf bE = load_b_gl(W0, 4 * DD, nt * 16, DD + kk * 32);
      a0 = wmma_bf16(load_a_lds(&Ebuf[ 0 * LDE + kk * 32], LDE), bE, a0);
      a1 = wmma_bf16(load_a_lds(&Ebuf[16 * LDE + kk * 32], LDE), bE, a1);
      a2 = wmma_bf16(load_a_lds(&Ebuf[32 * LDE + kk * 32], LDE), bE, a2);
      v16bf bG = load_b_gl(W0, 4 * DD, nt * 16, 3 * DD + kk * 32);
      a0 = wmma_bf16(load_a_lds(&Gbuf[ 0 * LDE + kk * 32], LDE), bG, a0);
      a1 = wmma_bf16(load_a_lds(&Gbuf[16 * LDE + kk * 32], LDE), bG, a1);
      a2 = wmma_bf16(load_a_lds(&Gbuf[32 * LDE + kk * 32], LDE), bG, a2);
    }
    float hv = hw0[colB];
    for (int r = 0; r < 8; ++r) {
      M0buf[( 0 + r + mo) * LDE + colB] = f2bf(gelu_exact(a0[r] + hv));
      M0buf[(16 + r + mo) * LDE + colB] = f2bf(gelu_exact(a1[r] + hv));
      M0buf[(32 + r + mo) * LDE + colB] = f2bf(gelu_exact(a2[r] + hv));
    }
  }
  __syncthreads();

  // m1 = gelu(m0 @ W1^T + b1)
  {
    v8f a0 = {}, a1 = {}, a2 = {};
    for (int kk = 0; kk < 4; ++kk) {
      v16bf b = load_b_gl(W1, DD, nt * 16, kk * 32);
      a0 = wmma_bf16(load_a_lds(&M0buf[ 0 * LDE + kk * 32], LDE), b, a0);
      a1 = wmma_bf16(load_a_lds(&M0buf[16 * LDE + kk * 32], LDE), b, a1);
      a2 = wmma_bf16(load_a_lds(&M0buf[32 * LDE + kk * 32], LDE), b, a2);
    }
    float bias = b1[colB];
    for (int r = 0; r < 8; ++r) {
      M1buf[( 0 + r + mo) * LDE + colB] = f2bf(gelu_exact(a0[r] + bias));
      M1buf[(16 + r + mo) * LDE + colB] = f2bf(gelu_exact(a1[r] + bias));
      M1buf[(32 + r + mo) * LDE + colB] = f2bf(gelu_exact(a2[r] + bias));
    }
  }
  __syncthreads();

  // m2 = m1 @ W2^T (+ b2 folded in later); column-sum all 48 rows
  {
    v8f a0 = {}, a1 = {}, a2 = {};
    for (int kk = 0; kk < 4; ++kk) {
      v16bf b = load_b_gl(W2, DD, nt * 16, kk * 32);
      a0 = wmma_bf16(load_a_lds(&M1buf[ 0 * LDE + kk * 32], LDE), b, a0);
      a1 = wmma_bf16(load_a_lds(&M1buf[16 * LDE + kk * 32], LDE), b, a1);
      a2 = wmma_bf16(load_a_lds(&M1buf[32 * LDE + kk * 32], LDE), b, a2);
    }
    float s = 0.f;
    for (int r = 0; r < 8; ++r) s += a0[r] + a1[r] + a2[r];
    sums[t] = s;                 // lane-local partial column sum
  }
  __syncthreads();
  // Deterministic cross-lane combine: col j==t for t<128; lanes l and l+16 of
  // wave t>>4 hold the two row-halves of column t.
  if (t < DD) {
    float total = sums[(t >> 4) * 32 + (t & 15)] +
                  sums[(t >> 4) * 32 + (t & 15) + 16];
    agg[(size_t)n * DD + t] = (total + (float)KK * b2[t]) * SCALE_INV;
  }
}

// ---------------------------------------------------------------------------
// Update kernel: 16 nodes per workgroup.
// h = LN1(h+agg); h = LN2(h + dense1(gelu(dense0(h)))) * mask
// ---------------------------------------------------------------------------
__global__ __launch_bounds__(256)
void update_kernel(float* __restrict__ hbuf,
                   const float* __restrict__ agg,
                   const float* __restrict__ mask,
                   const float* __restrict__ ln1w, const float* __restrict__ ln1b,
                   const unsigned short* __restrict__ Wd0,  // (512,128) bf16
                   const float* __restrict__ bd0,
                   const unsigned short* __restrict__ Wd1,  // (128,512) bf16
                   const float* __restrict__ bd1,
                   const float* __restrict__ ln2w, const float* __restrict__ ln2b) {
  __shared__ float Yf[16 * 132];
  __shared__ unsigned short Xbf[16 * LDE];
  __shared__ unsigned short Hbf[16 * LDH];
  __shared__ float rsum[256], rsq[256];
  __shared__ float mus[16], rstd[16];

  const int t    = threadIdx.x;
  const int lane = t & 31;
  const int wv   = t >> 5;
  const int nb   = blockIdx.x * 16;
  const int r    = t >> 4;      // row 0..15
  const int j    = t & 15;      // 16 threads/row, 8 cols each

  // y = h + agg ; LayerNorm1
  float yv[8]; float s = 0.f, s2 = 0.f;
  for (int e = 0; e < 8; ++e) {
    int c = j * 8 + e;
    float v = hbuf[(size_t)(nb + r) * DD + c] + agg[(size_t)(nb + r) * DD + c];
    yv[e] = v; s += v; s2 += v * v;
  }
  rsum[t] = s; rsq[t] = s2;
  __syncthreads();
  if (t < 16) {
    float a = 0.f, b = 0.f;
    for (int q = 0; q < 16; ++q) { a += rsum[t * 16 + q]; b += rsq[t * 16 + q]; }
    float mu = a / (float)DD;
    mus[t] = mu;
    rstd[t] = rsqrtf(b / (float)DD - mu * mu + LN_EPS);
  }
  __syncthreads();
  {
    float mu = mus[r], rs = rstd[r];
    for (int e = 0; e < 8; ++e) {
      int c = j * 8 + e;
      float hn = (yv[e] - mu) * rs * ln1w[c] + ln1b[c];
      Yf[r * 132 + c] = hn;            // fp32 copy for residual
      Xbf[r * LDE + c] = f2bf(hn);     // bf16 for WMMA
    }
  }
  __syncthreads();

  // hidden = gelu(X @ Wd0^T + bd0): (16,128)x(128,512)
  {
    v16bf afr[4];
    for (int kk = 0; kk < 4; ++kk) afr[kk] = load_a_lds(&Xbf[kk * 32], LDE);
    for (int nt = wv; nt < 32; nt += 8) {
      v8f acc = {};
      for (int kk = 0; kk < 4; ++kk)
        acc = wmma_bf16(afr[kk], load_b_gl(Wd0, DD, nt * 16, kk * 32), acc);
      int col = nt * 16 + (lane & 15);
      int mo  = (lane >> 4) << 3;
      float bias = bd0[col];
      for (int rr = 0; rr < 8; ++rr)
        Hbf[(rr + mo) * LDH + col] = f2bf(gelu_exact(acc[rr] + bias));
    }
  }
  __syncthreads();

  // d = H @ Wd1^T + bd1: (16,512)x(512,128); y2 = h1 + d (in place in Yf)
  {
    const int nt = wv;
    v8f acc = {};
    for (int kk = 0; kk < 16; ++kk)
      acc = wmma_bf16(load_a_lds(&Hbf[kk * 32], LDH),
                      load_b_gl(Wd1, DFF, nt * 16, kk * 32), acc);
    int col = nt * 16 + (lane & 15);
    int mo  = (lane >> 4) << 3;
    float bias = bd1[col];
    for (int rr = 0; rr < 8; ++rr)
      Yf[(rr + mo) * 132 + col] += acc[rr] + bias;
  }
  __syncthreads();

  // LayerNorm2 + mask + writeback
  {
    float a = 0.f, b = 0.f;
    for (int e = 0; e < 8; ++e) {
      float v = Yf[r * 132 + j * 8 + e];
      a += v; b += v * v;
    }
    rsum[t] = a; rsq[t] = b;
  }
  __syncthreads();
  if (t < 16) {
    float a = 0.f, b = 0.f;
    for (int q = 0; q < 16; ++q) { a += rsum[t * 16 + q]; b += rsq[t * 16 + q]; }
    float mu = a / (float)DD;
    mus[t] = mu;
    rstd[t] = rsqrtf(b / (float)DD - mu * mu + LN_EPS);
  }
  __syncthreads();
  {
    float mu = mus[r], rs = rstd[r], mk = mask[nb + r];
    for (int e = 0; e < 8; ++e) {
      int c = j * 8 + e;
      float v = ((Yf[r * 132 + c] - mu) * rs * ln2w[c] + ln2b[c]) * mk;
      hbuf[(size_t)(nb + r) * DD + c] = v;
    }
  }
}

// ---------------------------------------------------------------------------
extern "C" void kernel_launch(void* const* d_in, const int* in_sizes, int n_in,
                              void* d_out, int out_size, void* d_ws, size_t ws_size,
                              hipStream_t stream) {
  const float* node_f = (const float*)d_in[0];
  const float* edge_f = (const float*)d_in[1];
  const int*   nbr    = (const int*)d_in[2];
  const float* mask   = (const float*)d_in[3];
  const float* msg_w0 = (const float*)d_in[4];
  const float* msg_b0 = (const float*)d_in[5];
  const float* msg_w1 = (const float*)d_in[6];
  const float* msg_b1 = (const float*)d_in[7];
  const float* msg_w2 = (const float*)d_in[8];
  const float* msg_b2 = (const float*)d_in[9];
  const float* ln1w   = (const float*)d_in[10];
  const float* ln1b   = (const float*)d_in[11];
  const float* dw0    = (const float*)d_in[12];
  const float* db0    = (const float*)d_in[13];
  const float* dw1    = (const float*)d_in[14];
  const float* db1    = (const float*)d_in[15];
  const float* ln2w   = (const float*)d_in[16];
  const float* ln2b   = (const float*)d_in[17];
  (void)in_sizes; (void)n_in; (void)out_size; (void)ws_size;

  // d_out doubles as the evolving h buffer (final layer leaves the answer).
  float* h_buf = (float*)d_out;

  char* ws = (char*)d_ws;
  size_t off = 0;
  auto carve = [&](size_t bytes) -> char* {
    char* p = ws + off;
    off = (off + bytes + 255) & ~(size_t)255;
    return p;
  };
  float*          agg      = (float*)carve((size_t)NN * DD * 4);
  unsigned short* nodes_bf = (unsigned short*)carve((size_t)NN * DD * 2);
  unsigned short* w0b  = (unsigned short*)carve((size_t)LL * DD * 4 * DD * 2);
  unsigned short* w1b  = (unsigned short*)carve((size_t)LL * DD * DD * 2);
  unsigned short* w2b  = (unsigned short*)carve((size_t)LL * DD * DD * 2);
  unsigned short* wd0b = (unsigned short*)carve((size_t)LL * DFF * DD * 2);
  unsigned short* wd1b = (unsigned short*)carve((size_t)LL * DD * DFF * 2);

  auto cvt = [&](const float* s, unsigned short* d, int n) {
    k_cvt_bf16<<<(n + 255) / 256, 256, 0, stream>>>(s, d, n);
  };
  cvt(node_f, nodes_bf, NN * DD);
  cvt(msg_w0, w0b, LL * DD * 4 * DD);
  cvt(msg_w1, w1b, LL * DD * DD);
  cvt(msg_w2, w2b, LL * DD * DD);
  cvt(dw0,    wd0b, LL * DFF * DD);
  cvt(dw1,    wd1b, LL * DD * DFF);
  k_copy_f32<<<(NN * DD + 255) / 256, 256, 0, stream>>>(node_f, h_buf, NN * DD);

  for (int i = 0; i < LL; ++i) {
    msg_kernel<<<NN, 256, 0, stream>>>(
        h_buf, edge_f, nbr, nodes_bf,
        w0b + (size_t)i * DD * 4 * DD, msg_b0 + i * DD,
        w1b + (size_t)i * DD * DD,     msg_b1 + i * DD,
        w2b + (size_t)i * DD * DD,     msg_b2 + i * DD,
        agg);
    update_kernel<<<NN / 16, 256, 0, stream>>>(
        h_buf, agg, mask,
        ln1w + i * DD, ln1b + i * DD,
        wd0b + (size_t)i * DFF * DD, db0 + i * DFF,
        wd1b + (size_t)i * DD * DFF, db1 + i * DD,
        ln2w + i * DD, ln2b + i * DD);
  }
}
